// MultiHeadAttentionLayer_59004260713073
// MI455X (gfx1250) — matrix operand order repaired and checked by reading, not compiled, not run on hardware
//
#include <hip/hip_runtime.h>

// ---------------------------------------------------------------------------
// Transformer block for MI455X (gfx1250, wave32, WMMA + Tensor Data Mover).
//   S=4096, E=2048, F=8192, H=16, D=128
// GEMMs: TDM (tensor_load_to_lds) double-buffered tile staging, all math via
// v_wmma_f32_16x16x32_bf16. Attention: flash (online softmax), WMMA for both
// Q*K^T and P*V, fragments loaded straight from global memory.
// ---------------------------------------------------------------------------

typedef __attribute__((ext_vector_type(16))) __bf16 v16bf;
typedef __attribute__((ext_vector_type(8)))  float  v8f;
typedef __attribute__((ext_vector_type(4)))  unsigned int v4u;
typedef __attribute__((ext_vector_type(8)))  int  v8i;
typedef __attribute__((ext_vector_type(4)))  int  v4i;

union FragBF {
    uint4 q[2];
    v16bf v;
};
static_assert(sizeof(FragBF) == 32, "frag size");

static constexpr int S_ = 4096;
static constexpr int E_ = 2048;
static constexpr int F_ = 8192;
static constexpr int H_ = 16;
static constexpr int D_ = 128;

__device__ __forceinline__ unsigned short f32_to_bf16(float f) {
    unsigned int u = __builtin_bit_cast(unsigned int, f);
    unsigned int r = u + 0x7FFFu + ((u >> 16) & 1u);   // round-to-nearest-even
    return (unsigned short)(r >> 16);
}

__device__ __forceinline__ v8f wmma_bf16(const FragBF& a, const FragBF& b, v8f c) {
    return __builtin_amdgcn_wmma_f32_16x16x32_bf16(
        /*neg_a=*/false, a.v, /*neg_b=*/false, b.v,
        /*c_mod=*/(short)0, c, /*reuse_a=*/false, /*reuse_b=*/false);
}

// ---------------------------------------------------------------------------
// TDM: DMA one 2D bf16 tile (tile_d0 x tile_d1, row stride `stride` elems)
// from global memory into LDS (row-major, contiguous). D# per ISA 8.3-8.6.
// EXEC is ignored by TENSOR_* ops -> call from exactly one wave per tile.
// This toolchain exposes the 6-arg builtin:
//   (uint32x4 g0, int32x8 g1, int32x4 g2, int32x4 g3, int32x8 pad, i32 cpol)
// ---------------------------------------------------------------------------
__device__ __forceinline__ void tdm_load_tile_bf16(
    unsigned lds_off, const unsigned short* gsrc,
    unsigned tensor_d0, unsigned tensor_d1,
    unsigned tile_d0, unsigned tile_d1, unsigned stride) {
    unsigned long long ga = (unsigned long long)(size_t)gsrc;
    v4u g0;
    g0[0] = 1u;                                    // count=1 (valid user D#)
    g0[1] = lds_off;                               // lds_addr (bytes)
    g0[2] = (unsigned)(ga & 0xFFFFFFFFu);          // global_addr[31:0]
    g0[3] = (unsigned)((ga >> 32) & 0x01FFFFFFu)   // global_addr[56:32]
            | (2u << 30);                          // type=2 ("image")
    v8i g1;
    g1[0] = (int)(1u << 16);                       // data_size=1 (2 bytes)
    g1[1] = (int)((tensor_d0 & 0xFFFFu) << 16);    // tensor_dim0[15:0]
    g1[2] = (int)((tensor_d0 >> 16) | ((tensor_d1 & 0xFFFFu) << 16));
    g1[3] = (int)((tensor_d1 >> 16) | (tile_d0 << 16));   // tile_dim0
    g1[4] = (int)(tile_d1 & 0xFFFFu);              // tile_dim1 (tile_dim2=0)
    g1[5] = (int)stride;                           // tensor_dim0_stride[31:0]
    g1[6] = 0;                                     // stride hi, dim1_stride lo
    g1[7] = 0;
    v4i z4 = {0, 0, 0, 0};                         // groups 2/3 unused (2D)
    v8i z8 = {0, 0, 0, 0, 0, 0, 0, 0};
    __builtin_amdgcn_tensor_load_to_lds(g0, g1, z4, z4, z8, 0);
}

// ---------------------------------------------------------------------------
// f32 -> bf16 conversion (vectorized, n multiple of 4)
// ---------------------------------------------------------------------------
__global__ void cvt_f32_to_bf16(const float* __restrict__ src,
                                unsigned short* __restrict__ dst, int n4) {
    int i      = blockIdx.x * blockDim.x + threadIdx.x;
    int stride = gridDim.x * blockDim.x;
    const float4* s4 = (const float4*)src;
    uint2*        d4 = (uint2*)dst;
    for (; i < n4; i += stride) {
        float4 v = s4[i];
        uint2 o;
        o.x = (unsigned)f32_to_bf16(v.x) | ((unsigned)f32_to_bf16(v.y) << 16);
        o.y = (unsigned)f32_to_bf16(v.z) | ((unsigned)f32_to_bf16(v.w) << 16);
        d4[i] = o;
    }
}

// ---------------------------------------------------------------------------
// f32[R][C] -> bf16[C][R] transposed convert (weights, read exactly once).
// Storing W^T makes the GEMM B-tile a row-major rectangle (row stride K),
// so a single TDM descriptor stages it and consecutive-k pairs land exactly
// in the packed layout the WMMA B-fragment ds_load_b128s expect.
// ---------------------------------------------------------------------------
__global__ void cvt_transpose_f32_to_bf16(const float* __restrict__ src,
                                          unsigned short* __restrict__ dst,
                                          int R, int C) {
    __shared__ float tile[32][33];
    const int c0 = blockIdx.x * 32, r0 = blockIdx.y * 32;
#pragma unroll
    for (int j = 0; j < 4; ++j) {
        int r = threadIdx.y + j * 8;
        tile[r][threadIdx.x] = src[(size_t)(r0 + r) * C + c0 + threadIdx.x];
    }
    __syncthreads();
#pragma unroll
    for (int j = 0; j < 4; ++j) {
        int c = threadIdx.y + j * 8;
        dst[(size_t)(c0 + c) * R + r0 + threadIdx.x] =
            f32_to_bf16(tile[threadIdx.x][c]);
    }
}

// ---------------------------------------------------------------------------
// bf16 GEMM:  C[M,N] = A[M,K] * Bt[N,K]^T + bias   (optional ReLU)
//   - 128x128 block tile, 256 threads (8 waves), K-step 32
//   - A tile (128x32) and Bt tile (128x32) staged by TDM, double-buffered:
//     wave0/wave1 issue tensor_load_to_lds for iter i+1 while all waves run
//     WMMAs on iter i. s_wait_tensorcnt + barrier at loop head.
//   - TRANS_OUT stores bf16 C transposed (produces V^T directly)
// Requires M,N,K multiples of 128 (true for all call sites).
// ---------------------------------------------------------------------------
template <bool RELU, bool OUT_BF16, bool TRANS_OUT>
__global__ __launch_bounds__(256) void gemm_bf16(
    const unsigned short* __restrict__ A,
    const unsigned short* __restrict__ Bt,
    const float* __restrict__ bias,
    void* __restrict__ Cout,
    int M, int N, int K, int ldT) {
    __shared__ __align__(16) unsigned int aP[2][128 * 16];  // 2 x 8 KB
    __shared__ __align__(16) unsigned int bP[2][128 * 16];  // 2 x 8 KB

    const int tid  = threadIdx.x;
    const int lane = tid & 31;
    const int wid  = tid >> 5;
    const int wm   = wid & 1;        // wave row group (64 rows)
    const int wn   = wid >> 1;       // wave col group (32 cols)
    const int lm   = lane & 15;
    const int lh   = lane >> 4;
    const int m0   = blockIdx.y * 128;
    const int n0   = blockIdx.x * 128;

    const unsigned short* Abase  = A  + (size_t)m0 * K;
    const unsigned short* Btbase = Bt + (size_t)n0 * K;
    const unsigned aOff0 = (unsigned)(size_t)&aP[0][0];
    const unsigned aOff1 = (unsigned)(size_t)&aP[1][0];
    const unsigned bOff0 = (unsigned)(size_t)&bP[0][0];
    const unsigned bOff1 = (unsigned)(size_t)&bP[1][0];

    v8f acc[4][2] = {};

    const int niter = K >> 5;
    // prologue: stage tiles for iteration 0 into buffer 0
    if (wid == 0)
        tdm_load_tile_bf16(aOff0, Abase, 32, 128, 32, 128, (unsigned)K);
    else if (wid == 1)
        tdm_load_tile_bf16(bOff0, Btbase, 32, 128, 32, 128, (unsigned)K);

    for (int i = 0; i < niter; ++i) {
        const int b = i & 1;
        if (wid < 2) __builtin_amdgcn_s_wait_tensorcnt(0);
        __syncthreads();  // tiles for iter i visible; iter i-1 reads done

        if (i + 1 < niter) {  // prefetch next tiles into the other buffer
            const int k1 = (i + 1) << 5;
            if (wid == 0)
                tdm_load_tile_bf16(b ? aOff0 : aOff1, Abase + k1,
                                   32, 128, 32, 128, (unsigned)K);
            else if (wid == 1)
                tdm_load_tile_bf16(b ? bOff0 : bOff1, Btbase + k1,
                                   32, 128, 32, 128, (unsigned)K);
        }

        // ---- fragments (2x ds_load_b128 each) + 8 WMMAs ----
        FragBF af[4], bfrag[2];
#pragma unroll
        for (int mi = 0; mi < 4; ++mi) {
            int m = wm * 64 + mi * 16 + lm;
            af[mi].q[0] = *(const uint4*)&aP[b][m * 16 + 4 * lh];
            af[mi].q[1] = *(const uint4*)&aP[b][m * 16 + 8 + 4 * lh];
        }
#pragma unroll
        for (int ni = 0; ni < 2; ++ni) {
            int n = wn * 32 + ni * 16 + lm;
            bfrag[ni].q[0] = *(const uint4*)&bP[b][n * 16 + 4 * lh];
            bfrag[ni].q[1] = *(const uint4*)&bP[b][n * 16 + 8 + 4 * lh];
        }
#pragma unroll
        for (int mi = 0; mi < 4; ++mi)
#pragma unroll
            for (int ni = 0; ni < 2; ++ni)
                acc[mi][ni] = wmma_bf16(af[mi], bfrag[ni], acc[mi][ni]);
    }

    // ---- epilogue: bias (+ReLU) + store ----
#pragma unroll
    for (int ni = 0; ni < 2; ++ni) {
        int col  = n0 + wn * 32 + ni * 16 + lm;
        float bv = bias[col];
#pragma unroll
        for (int mi = 0; mi < 4; ++mi) {
#pragma unroll
            for (int r = 0; r < 8; ++r) {
                int row = m0 + wm * 64 + mi * 16 + r + 8 * lh;
                float v = acc[mi][ni][r] + bv;
                if (RELU) v = fmaxf(v, 0.0f);
                if (OUT_BF16) {
                    unsigned short* o = (unsigned short*)Cout;
                    if (TRANS_OUT)
                        o[(size_t)col * ldT + row] = f32_to_bf16(v);
                    else
                        o[(size_t)row * N + col] = f32_to_bf16(v);
                } else {
                    ((float*)Cout)[(size_t)row * N + col] = v;
                }
            }
        }
    }
}

// ---------------------------------------------------------------------------
// Flash attention (causal), one wave per (head, 16-query-row) tile.
//   Q,K: bf16 [S][E] row-major (head h at column h*D)
//   Vt : bf16 [E][S]  (V transposed; written by the V-projection GEMM)
//   O  : bf16 [S][E]
// Per 32-key block: 8 WMMAs for Q*K^T, 8 WMMAs for P*V.
// ---------------------------------------------------------------------------
__global__ __launch_bounds__(128) void flash_attn_kernel(
    const unsigned short* __restrict__ Q,
    const unsigned short* __restrict__ Kmat,
    const unsigned short* __restrict__ Vt,
    unsigned short* __restrict__ O) {
    __shared__ __align__(16) unsigned short pbuf[4][16 * 32];  // P tile per wave

    const int tid  = threadIdx.x;
    const int lane = tid & 31;
    const int w    = tid >> 5;
    const int lm   = lane & 15;
    const int lh   = lane >> 4;
    const int h    = blockIdx.y;
    const int q0   = (blockIdx.x * 4 + w) * 16;
    const float scale = 0.08838834764831845f;  // 1/sqrt(128)

    // Q fragments (A-layout): 4 tiles of 16x32 covering D=128
    FragBF qf[4];
#pragma unroll
    for (int kt = 0; kt < 4; ++kt) {
        const unsigned short* base = Q + (size_t)(q0 + lm) * E_ + h * D_ + kt * 32;
        qf[kt].q[0] = *(const uint4*)(base + 8 * lh);
        qf[kt].q[1] = *(const uint4*)(base + 16 + 8 * lh);
    }

    v8f o[8] = {};
    float mrow[8], lrow[8];
#pragma unroll
    for (int r = 0; r < 8; ++r) {
        mrow[r] = -1e30f;
        lrow[r] = 0.0f;
    }

    const int jbmax = (q0 + 15) >> 5;
    for (int jb = 0; jb <= jbmax; ++jb) {
        const int kb0 = jb * 32;

        // ---- scores S = Q * K^T (16 x 32 tile, two 16x16 accumulators) ----
        v8f sa = {}, sb = {};
#pragma unroll
        for (int kt = 0; kt < 4; ++kt) {
            FragBF k0f, k1f;
            const unsigned short* kb =
                Kmat + (size_t)(kb0 + lm) * E_ + h * D_ + kt * 32;
            k0f.q[0] = *(const uint4*)(kb + 8 * lh);
            k0f.q[1] = *(const uint4*)(kb + 16 + 8 * lh);
            const unsigned short* kb1 = kb + (size_t)16 * E_;
            k1f.q[0] = *(const uint4*)(kb1 + 8 * lh);
            k1f.q[1] = *(const uint4*)(kb1 + 16 + 8 * lh);
            sa = wmma_bf16(qf[kt], k0f, sa);
            sb = wmma_bf16(qf[kt], k1f, sb);
        }

        // ---- causal mask + online softmax update ----
#pragma unroll
        for (int r = 0; r < 8; ++r) {
            const int row = q0 + r + 8 * lh;
            const int c0  = kb0 + lm;
            float s0 = sa[r] * scale;
            if (c0 > row) s0 = -1e30f;
            float s1 = sb[r] * scale;
            if (c0 + 16 > row) s1 = -1e30f;

            float mx = fmaxf(s0, s1);
#pragma unroll
            for (int off = 8; off >= 1; off >>= 1)
                mx = fmaxf(mx, __shfl_xor(mx, off, 32));
            float mnew  = fmaxf(mrow[r], mx);
            float alpha = __expf(mrow[r] - mnew);
            float p0    = __expf(s0 - mnew);
            float p1    = __expf(s1 - mnew);
            float ps    = p0 + p1;
#pragma unroll
            for (int off = 8; off >= 1; off >>= 1)
                ps += __shfl_xor(ps, off, 32);
            lrow[r] = lrow[r] * alpha + ps;
            mrow[r] = mnew;
#pragma unroll
            for (int t = 0; t < 8; ++t) o[t][r] *= alpha;

            // P (C-layout) -> LDS row-major 16x32 bf16 (== packed A layout)
            pbuf[w][(r + 8 * lh) * 32 + lm]      = f32_to_bf16(p0);
            pbuf[w][(r + 8 * lh) * 32 + 16 + lm] = f32_to_bf16(p1);
        }

        // per-wave LDS relayout: make stores visible before fragment reads
        asm volatile("s_wait_dscnt 0" ::: "memory");

        FragBF pf;
        const unsigned short* pb = &pbuf[w][lm * 32];
        pf.q[0] = *(const uint4*)(pb + 8 * lh);
        pf.q[1] = *(const uint4*)(pb + 16 + 8 * lh);

        // ---- O += P * V  (V^T rows are contiguous in key -> direct B frags)
#pragma unroll
        for (int t = 0; t < 8; ++t) {
            FragBF vf;
            const unsigned short* vb =
                Vt + (size_t)(h * D_ + t * 16 + lm) * S_ + kb0 + 8 * lh;
            vf.q[0] = *(const uint4*)vb;
            vf.q[1] = *(const uint4*)(vb + 16);
            o[t] = wmma_bf16(pf, vf, o[t]);
        }
    }

    // ---- normalize + store ----
#pragma unroll
    for (int r = 0; r < 8; ++r) {
        float inv     = 1.0f / lrow[r];
        const int row = q0 + r + 8 * lh;
#pragma unroll
        for (int t = 0; t < 8; ++t)
            O[(size_t)row * E_ + h * D_ + t * 16 + lm] =
                f32_to_bf16(o[t][r] * inv);
    }
}

// ---------------------------------------------------------------------------
// Host-side orchestration
// ---------------------------------------------------------------------------
extern "C" void kernel_launch(void* const* d_in, const int* in_sizes, int n_in,
                              void* d_out, int out_size, void* d_ws,
                              size_t ws_size, hipStream_t stream) {
    (void)in_sizes; (void)n_in; (void)out_size; (void)ws_size;

    const float* emb = (const float*)d_in[0];
    const float* Wq  = (const float*)d_in[1];
    const float* bq  = (const float*)d_in[2];
    const float* Wk  = (const float*)d_in[3];
    const float* bk  = (const float*)d_in[4];
    const float* Wv  = (const float*)d_in[5];
    const float* bv  = (const float*)d_in[6];
    const float* W1  = (const float*)d_in[7];
    const float* b1  = (const float*)d_in[8];
    const float* W2  = (const float*)d_in[9];
    const float* b2  = (const float*)d_in[10];
    float* out = (float*)d_out;

    // Workspace layout (bf16 buffers), stream-ordered reuse:
    //   [0,16MB)      emb_bf  -> later attn_bf
    //   [16MB,88MB)   WqT|WkT|WvT|q|k|vt  -> later hidden (64MB)
    //   [88MB..)      W1T -> later W2T
    char* ws = (char*)d_ws;
    unsigned short* embbf  = (unsigned short*)ws;
    unsigned short* attnbf = embbf;
    char* regB = ws + (size_t)S_ * E_ * 2;                 // +16 MiB
    unsigned short* wqbf = (unsigned short*)(regB);
    unsigned short* wkbf = (unsigned short*)(regB + (size_t)8388608);
    unsigned short* wvbf = (unsigned short*)(regB + (size_t)16777216);
    unsigned short* qbf  = (unsigned short*)(regB + (size_t)25165824);
    unsigned short* kbf  = (unsigned short*)(regB + (size_t)41943040);
    unsigned short* vtbf = (unsigned short*)(regB + (size_t)58720256);
    unsigned short* hidbf = (unsigned short*)(regB);       // 64 MiB overlay
    unsigned short* w1bf = (unsigned short*)(ws + (size_t)92274688);
    unsigned short* w2bf = w1bf;                           // overlay after FFN1

    dim3 tb(32, 8);

    // 1) bf16 activation convert + transposed bf16 weight converts
    cvt_f32_to_bf16<<<2048, 256, 0, stream>>>(emb, embbf, (S_ * E_) / 4);
    cvt_transpose_f32_to_bf16<<<dim3(E_ / 32, E_ / 32), tb, 0, stream>>>(
        Wq, wqbf, E_, E_);
    cvt_transpose_f32_to_bf16<<<dim3(E_ / 32, E_ / 32), tb, 0, stream>>>(
        Wk, wkbf, E_, E_);
    cvt_transpose_f32_to_bf16<<<dim3(E_ / 32, E_ / 32), tb, 0, stream>>>(
        Wv, wvbf, E_, E_);

    // 2) QKV projections (V stored transposed as V^T [E][S])
    dim3 gProj(E_ / 128, S_ / 128);
    gemm_bf16<false, true, false><<<gProj, 256, 0, stream>>>(
        embbf, wqbf, bq, qbf, S_, E_, E_, 0);
    gemm_bf16<false, true, false><<<gProj, 256, 0, stream>>>(
        embbf, wkbf, bk, kbf, S_, E_, E_, 0);
    gemm_bf16<false, true, true><<<gProj, 256, 0, stream>>>(
        embbf, wvbf, bv, vtbf, S_, E_, E_, /*ldT=*/S_);

    // 3) causal flash attention -> attn_bf (reuses emb_bf region)
    flash_attn_kernel<<<dim3(S_ / 64, H_), 128, 0, stream>>>(
        qbf, kbf, vtbf, attnbf);

    // 4) FFN1: hidden = relu(attn @ W1 + b1)
    cvt_transpose_f32_to_bf16<<<dim3(F_ / 32, E_ / 32), tb, 0, stream>>>(
        W1, w1bf, E_, F_);
    gemm_bf16<true, true, false><<<dim3(F_ / 128, S_ / 128), 256, 0, stream>>>(
        attnbf, w1bf, b1, hidbf, S_, F_, E_, 0);

    // 5) FFN2: out = hidden @ W2 + b2   (f32 output)
    cvt_transpose_f32_to_bf16<<<dim3(E_ / 32, F_ / 32), tb, 0, stream>>>(
        W2, w2bf, F_, E_);
    gemm_bf16<false, false, false><<<dim3(E_ / 128, S_ / 128), 256, 0, stream>>>(
        hidbf, w2bf, b2, out, S_, E_, F_, 0);
}
